// NaviEncoder_52355651338944
// MI455X (gfx1250) — matrix operand order repaired and checked by reading, not compiled, not run on hardware
//
#include <hip/hip_runtime.h>
#include <hip/hip_bf16.h>

// Problem constants (from reference)
#define N_SC 64
#define N_AG 1024
#define N_MP 1024
#define HID  256
#define KTOT 512          // fused K = HIDDEN (gathered) + PE_DIM (pe)
#define AGB  64           // agents per block (M panel)
#define MT   4            // 16-row M tiles per block
#define NT   16           // 16-col N tiles (HID/16)
#define KB   16           // K blocks of 32 (KTOT/32)

typedef __attribute__((ext_vector_type(16))) __bf16 v16bf;
typedef __attribute__((ext_vector_type(8)))  float  v8f;

// round-to-nearest-even f32 -> bf16
__device__ __forceinline__ unsigned short f2bf(float x) {
  unsigned u = __float_as_uint(x);
  u += 0x7FFFu + ((u >> 16) & 1u);
  return (unsigned short)(u >> 16);
}
__device__ __forceinline__ unsigned pack2(float a, float b) {
  return (unsigned)f2bf(a) | ((unsigned)f2bf(b) << 16);
}

// A-fragment address (in halves) for element (row a, depth k), per ISA 16-bit
// A-matrix 16x32 layout: lanes 0-15 hold K {0..7,16..23}, lanes 16-31 hold
// K {8..15,24..31}; halves packed sequentially.
__device__ __forceinline__ int aAddr(int a, int k) {
  int mt = a >> 4, m = a & 15;
  int kb = k >> 5, kr = k & 31;
  int lane = m + (((kr >> 3) & 1) << 4);
  int hi   = (kr & 7) + ((kr & 16) >> 1);
  return ((mt * KB + kb) * 32 + lane) * 16 + hi;
}

// ---------------------------------------------------------------------------
// Prep: stacked [W_mp ; W_pe] (512 x 256) f32 -> bf16 B-fragments in d_ws.
// B-fragment layout: tile (kb, nt); lane n (0-15) holds column N = nt*16+n,
// K = kb*32 + {0..7,16..23} as 16 sequential halves; lane n+16 holds K = kb*32+{8..15,24..31}.
// ---------------------------------------------------------------------------
__global__ __launch_bounds__(256) void prep_wfrag(
    const float* __restrict__ W_mp, const float* __restrict__ W_pe,
    unsigned short* __restrict__ Wfrag) {
  int f = blockIdx.x * blockDim.x + threadIdx.x;   // 0 .. 512*256-1
  int hi   = f & 15;
  int lane = (f >> 4) & 31;
  int nt   = (f >> 9) & 15;
  int kb   = f >> 13;
  int n = nt * 16 + (lane & 15);
  int k = kb * 32 + ((hi >> 3) << 4) + ((lane >> 4) << 3) + (hi & 7);
  float w = (k < HID) ? W_mp[k * HID + n] : W_pe[(k - HID) * HID + n];
  Wfrag[f] = f2bf(w);
}

// ---------------------------------------------------------------------------
// Main fused kernel: gather + pose/pe (VALU) + one K=512 bf16 WMMA GEMM.
// Block: 256 threads (8 wave32), 64 agents x 256 outputs. LDS = 64 KB A panel.
// Each thread owns agent a = tid&63 and a 64-wide channel strip (tid>>6)*64;
// the pose quad is recomputed per thread (4x redundant) so no pose LDS or
// extra barrier is needed.
// ---------------------------------------------------------------------------
__global__ __launch_bounds__(256) void navi_main(
    const int*   __restrict__ navi,   const float* __restrict__ ag_pose,
    const float* __restrict__ mpf,    const float* __restrict__ mpp,
    const float* __restrict__ b_mp,   const float* __restrict__ W_emb,
    const float* __restrict__ b_emb,  const float* __restrict__ b_pe,
    const unsigned short* __restrict__ Wfrag, float* __restrict__ out) {
  __shared__ __align__(32) unsigned short sA[MT * KB * 512]; // 64 KB bf16 A panel

  const int tid = threadIdx.x;
  const int b   = blockIdx.y;
  const int a0  = blockIdx.x * AGB;

  // ---- Per-thread agent / strip assignment --------------------------------
  const int a     = tid & (AGB - 1);       // agent within panel
  const int strip = (tid >> 6) * 64;       // 64-wide channel strip
  const int ga    = a0 + a;
  const int row   = navi[b * N_AG + ga];

  // ---- Pose math (redundant per thread; 4 threads/agent) ------------------
  float q0, q1, q2, q3;
  {
    const float* ap = ag_pose + (size_t)(b * N_AG + ga) * 3;
    const float* gp = mpp     + (size_t)(b * N_MP + row) * 3;
    float ax = ap[0], ay = ap[1], yaw = ap[2];
    float gx = gp[0], gy = gp[1], gyaw = gp[2];
    float c = cosf(yaw), s = sinf(yaw);
    float dx = gx - ax, dy = gy - ay;
    float yl = gyaw - yaw;
    q0 = dx * c + dy * s;
    q1 = dy * c - dx * s;
    q2 = cosf(yl);
    q3 = sinf(yl);
  }

  unsigned* sA32 = (unsigned*)sA;

  // ---- Gather features -> A panel K = 0..255 (bf16, fragment layout) ------
  {
    const float4* src =
        (const float4*)(mpf + ((size_t)b * N_MP + row) * HID + strip);
#pragma unroll
    for (int i = 0; i < 16; ++i) {
      float4 v = src[i];
      int k = strip + i * 4;
      sA32[aAddr(a, k)     >> 1] = pack2(v.x, v.y);
      sA32[aAddr(a, k + 2) >> 1] = pack2(v.z, v.w);
    }
  }

  // ---- pe = relu(pe_in @ W_emb + b_emb) -> A panel K = 256..511 -----------
  // W_emb/b_emb reads are lane-uniform per p (scalar-cache friendly).
  {
#pragma unroll
    for (int j = 0; j < 32; ++j) {
      int p = strip + j * 2;
      float v0 = q0 * W_emb[p]         + q1 * W_emb[HID + p] +
                 q2 * W_emb[2*HID + p] + q3 * W_emb[3*HID + p] + b_emb[p];
      float v1 = q0 * W_emb[p + 1]         + q1 * W_emb[HID + p + 1] +
                 q2 * W_emb[2*HID + p + 1] + q3 * W_emb[3*HID + p + 1] +
                 b_emb[p + 1];
      v0 = v0 > 0.f ? v0 : 0.f;
      v1 = v1 > 0.f ? v1 : 0.f;
      sA32[aAddr(a, HID + p) >> 1] = pack2(v0, v1);
    }
  }
  __syncthreads();

  // ---- GEMM: wave owns 2 N-tiles x 4 M-tiles; one B load feeds 4 WMMAs ----
  {
    int wave = tid >> 5, lane = tid & 31;
    const int ncol = lane & 15;
    const int rofs = (lane >> 4) << 3;         // +8 rows for lanes 16..31
    float* op = out + (size_t)b * N_AG * HID;
#pragma unroll
    for (int ntl = 0; ntl < 2; ++ntl) {
      int nt = wave * 2 + ntl;
      v8f acc0 = {}, acc1 = {}, acc2 = {}, acc3 = {};
#pragma unroll 4
      for (int kb = 0; kb < KB; ++kb) {
        const v16bf bf =
            *(const v16bf*)(Wfrag + (((size_t)(kb * NT + nt) * 32 + lane) << 4));
        v16bf af0 = *(const v16bf*)(sA + (((0 * KB + kb) * 32 + lane) << 4));
        v16bf af1 = *(const v16bf*)(sA + (((1 * KB + kb) * 32 + lane) << 4));
        v16bf af2 = *(const v16bf*)(sA + (((2 * KB + kb) * 32 + lane) << 4));
        v16bf af3 = *(const v16bf*)(sA + (((3 * KB + kb) * 32 + lane) << 4));
        acc0 = __builtin_amdgcn_wmma_f32_16x16x32_bf16(
            false, af0, false, bf, (short)0, acc0, false, false);
        acc1 = __builtin_amdgcn_wmma_f32_16x16x32_bf16(
            false, af1, false, bf, (short)0, acc1, false, false);
        acc2 = __builtin_amdgcn_wmma_f32_16x16x32_bf16(
            false, af2, false, bf, (short)0, acc2, false, false);
        acc3 = __builtin_amdgcn_wmma_f32_16x16x32_bf16(
            false, af3, false, bf, (short)0, acc3, false, false);
      }
      int n = nt * 16 + ncol;
      float bias = b_mp[n] + b_pe[n];
#pragma unroll
      for (int v = 0; v < 8; ++v) {
        // D layout: VGPR v -> row M = v (lanes 0-15) / v+8 (lanes 16-31)
        op[(size_t)(a0 +      v + rofs) * HID + n] = acc0[v] + bias;
        op[(size_t)(a0 + 16 + v + rofs) * HID + n] = acc1[v] + bias;
        op[(size_t)(a0 + 32 + v + rofs) * HID + n] = acc2[v] + bias;
        op[(size_t)(a0 + 48 + v + rofs) * HID + n] = acc3[v] + bias;
      }
    }
  }
}

extern "C" void kernel_launch(void* const* d_in, const int* in_sizes, int n_in,
                              void* d_out, int out_size, void* d_ws,
                              size_t ws_size, hipStream_t stream) {
  const int*   navi    = (const int*)d_in[0];
  const float* ag_pose = (const float*)d_in[1];
  const float* mpf     = (const float*)d_in[2];
  const float* mpp     = (const float*)d_in[3];
  const float* W_mp    = (const float*)d_in[4];
  const float* b_mp    = (const float*)d_in[5];
  const float* W_emb   = (const float*)d_in[6];
  const float* b_emb   = (const float*)d_in[7];
  const float* W_pe    = (const float*)d_in[8];
  const float* b_pe    = (const float*)d_in[9];
  unsigned short* Wfrag = (unsigned short*)d_ws;   // 512*256*2 = 256 KB
  float* outp = (float*)d_out;

  prep_wfrag<<<(KTOT * HID) / 256, 256, 0, stream>>>(W_mp, W_pe, Wfrag);
  navi_main<<<dim3(N_AG / AGB, N_SC), 256, 0, stream>>>(
      navi, ag_pose, mpf, mpp, b_mp, W_emb, b_emb, b_pe, Wfrag, outp);
}
